// KDEAttentionV2_69801808494920
// MI455X (gfx1250) — matrix operand order repaired and checked by reading, not compiled, not run on hardware
//
#include <hip/hip_runtime.h>
#include <hip/hip_bf16.h>

// ---------------------------------------------------------------------------
// KDE window attention, MI455X (gfx1250, wave32).
//   K1: TDM tile staging (TENSOR_LOAD_TO_LDS, 3D tile w/ LDS padding,
//       TENSORcnt), gram via V_WMMA_F32_16X16X32_F16, diag->sq,
//       exp row means -> density
//   K2: async-to-LDS density gather (GLOBAL_LOAD_ASYNC_TO_LDS_B32/ASYNCcnt),
//       MLP 128->32 relu ->128 sigmoid+gamma -> attn (4MB ws)
//   K3: fused bilinear upsample (align-corners) * x -> out   (BW bound)
// ---------------------------------------------------------------------------

typedef _Float16 v16h __attribute__((ext_vector_type(16)));
typedef float    v8f  __attribute__((ext_vector_type(8)));
typedef int      v4i  __attribute__((ext_vector_type(4)));
typedef int      v8i  __attribute__((ext_vector_type(8)));

#define F32_STRIDE 68   // 64 floats data + 4 floats TDM pad (272B row)

__device__ __forceinline__ void async_b32(unsigned int lds_addr, const float* gptr) {
  asm volatile("global_load_async_to_lds_b32 %0, %1, off"
               :: "v"(lds_addr), "v"(gptr) : "memory");
}

__device__ __forceinline__ v16h pack16(float4 x0, float4 x1, float4 x2, float4 x3) {
  v16h h;
  h[0]  = (_Float16)x0.x; h[1]  = (_Float16)x0.y; h[2]  = (_Float16)x0.z; h[3]  = (_Float16)x0.w;
  h[4]  = (_Float16)x1.x; h[5]  = (_Float16)x1.y; h[6]  = (_Float16)x1.z; h[7]  = (_Float16)x1.w;
  h[8]  = (_Float16)x2.x; h[9]  = (_Float16)x2.y; h[10] = (_Float16)x2.z; h[11] = (_Float16)x2.w;
  h[12] = (_Float16)x3.x; h[13] = (_Float16)x3.y; h[14] = (_Float16)x3.z; h[15] = (_Float16)x3.w;
  return h;
}

// ---------------------------------------------------------------------------
// Kernel 1: density. One wave per window-group. 4 waves / block (128 thr).
// n = ((b*4 + grp)*32 + hy)*32 + wx ; 32768 total window-groups.
// Tile: X=8 contig floats, Y=8 rows stride 256, Z=32 channels stride 65536.
// TDM writes 64 floats per channel + 16B pad -> stride-68 LDS rows.
// ---------------------------------------------------------------------------
__global__ __launch_bounds__(128) void kde_density_kernel(
    const float* __restrict__ x, float* __restrict__ density) {
  const int tid  = threadIdx.x;
  const int wave = tid >> 5;
  const int lane = tid & 31;
  const int lm   = lane & 15;
  const int hi   = lane >> 4;           // half-of-wave selector

  const int n   = blockIdx.x * 4 + wave;
  const int wx  = n & 31;
  const int hy  = (n >> 5) & 31;
  const int grp = (n >> 10) & 3;
  const int b   = n >> 12;

  __shared__ float tile[4][32 * F32_STRIDE];   // f32 tile, per wave (~8.7KB)
  __shared__ float sqs[4][32];                 // per-channel squared norms

  const size_t chan0 = (size_t)(b * 128 + grp * 32);
  const float* xbase = x + ((chan0 * 256 + (size_t)(hy * 8)) * 256 + (size_t)(wx * 8));
  float* tw = tile[wave];

  // ---- TDM: one tensor_load_to_lds per wave stages the whole 8KB tile -----
  {
    unsigned long long ga = (unsigned long long)(uintptr_t)xbase;
    unsigned int ldsb = (unsigned int)(uintptr_t)tw;
    v4i g0;
    g0[0] = 1;                                              // count=1 (valid user D#)
    g0[1] = __builtin_amdgcn_readfirstlane((int)ldsb);      // lds_addr (bytes)
    g0[2] = __builtin_amdgcn_readfirstlane((int)(unsigned int)ga);          // gaddr lo
    g0[3] = __builtin_amdgcn_readfirstlane(
              (int)(((unsigned int)(ga >> 32) & 0x01FFFFFFu) | 0x80000000u)); // gaddr hi | type=2
    v8i g1;
    g1[0] = 0x07520000;    // data_size=4B | pad_enable | pad_interval=64DW | pad_amount=4DW
    g1[1] = 0;             // atomic_barrier_addr=0 | tensor_dim0 lo16 (=0, dim0=1<<20)
    g1[2] = 0x00000010;    // tensor_dim0 hi16 (1<<20) | tensor_dim1 lo16 (=0)
    g1[3] = 0x00080010;    // tensor_dim1 hi16 (1<<20) | tile_dim0=8
    g1[4] = 0x00200008;    // tile_dim1=8 | tile_dim2=32
    g1[5] = 256;           // tensor_dim0_stride = 256 elements (row pitch)
    g1[6] = 0;             // dim0_stride hi | dim1_stride lo16
    g1[7] = 1;             // tensor_dim1_stride = 65536 elements (channel pitch)
    v4i g2;
    g2[0] = 1 << 20;       // tensor_dim2 (no OOB)
    g2[1] = 1;             // tensor_dim3
    g2[2] = 0;             // tensor_dim2_stride lo (unused: tile_dim3=0)
    g2[3] = 0;             // tile_dim3=0
    v4i g3;
    g3[0] = 0;             // tensor_dim3_stride lo
    g3[1] = 0x00010000;    // tensor_dim4=1 (bits 63:48)
    g3[2] = 0;             // tile_dim4=0
    g3[3] = 0;
    asm volatile("tensor_load_to_lds %0, %1, %2, %3"
                 :: "s"(g0), "s"(g1), "s"(g2), "s"(g3) : "memory");
    asm volatile("s_wait_tensorcnt 0x0" ::: "memory");
  }
  __syncthreads();

  // ---- gram = g * g^T via 2x2 tiles of v_wmma_f32_16x16x32_f16 -------------
  // A 16x32 f16 layout: lane lm (hi half) row = mi*16+lm,
  //   halves j=0..7 -> K = kb*32 + hi*8 + j ; j=8..15 -> K = kb*32 + 16 + hi*8 + (j-8)
  // B 32x16 f16 layout: lane lm col d = ni*16+lm, same K mapping as A
  v8f acc00 = {}, acc01 = {}, acc10 = {}, acc11 = {};
#pragma unroll
  for (int kb = 0; kb < 2; ++kb) {
    const float* pa0 = tw + (lm)      * F32_STRIDE + kb * 32 + hi * 8;
    const float* pa1 = tw + (16 + lm) * F32_STRIDE + kb * 32 + hi * 8;
    v16h A0 = pack16(*(const float4*)(pa0),      *(const float4*)(pa0 + 4),
                     *(const float4*)(pa0 + 16), *(const float4*)(pa0 + 20));
    v16h A1 = pack16(*(const float4*)(pa1),      *(const float4*)(pa1 + 4),
                     *(const float4*)(pa1 + 16), *(const float4*)(pa1 + 20));

    const float* pb0 = tw + (lm)      * F32_STRIDE + kb * 32 + hi * 8;
    const float* pb1 = tw + (16 + lm) * F32_STRIDE + kb * 32 + hi * 8;
    v16h B0 = pack16(*(const float4*)(pb0),     *(const float4*)(pb0 + 4),
                     *(const float4*)(pb0 + 16), *(const float4*)(pb0 + 20));
    v16h B1 = pack16(*(const float4*)(pb1),     *(const float4*)(pb1 + 4),
                     *(const float4*)(pb1 + 16), *(const float4*)(pb1 + 20));

    acc00 = __builtin_amdgcn_wmma_f32_16x16x32_f16(false, A0, false, B0, (short)0, acc00, false, false);
    acc01 = __builtin_amdgcn_wmma_f32_16x16x32_f16(false, A0, false, B1, (short)0, acc01, false, false);
    acc10 = __builtin_amdgcn_wmma_f32_16x16x32_f16(false, A1, false, B0, (short)0, acc10, false, false);
    acc11 = __builtin_amdgcn_wmma_f32_16x16x32_f16(false, A1, false, B1, (short)0, acc11, false, false);
  }

  // ---- sq = diag(gram). Tile(t,t): VGPR r holds row t*16 + hi*8 + r, col = t*16+lm.
  {
    const int r = lm - hi * 8;                       // valid when 0 <= r < 8
    const bool diagValid = (r >= 0) && (r < 8);
    float d0 = 0.f, d1 = 0.f;
#pragma unroll
    for (int rr = 0; rr < 8; ++rr) {
      if (rr == r) { d0 = acc00[rr]; d1 = acc11[rr]; }
    }
    if (diagValid) {
      sqs[wave][lm]      = d0;
      sqs[wave][16 + lm] = d1;
    }
  }
  __syncthreads();

  // ---- density rows: exp(-0.5*max(sq_c+sq_d-2*gram,0)), mean over d --------
  const float sqc0 = sqs[wave][lm];        // column ni=0
  const float sqc1 = sqs[wave][16 + lm];   // column ni=1
  float sqr[16];
#pragma unroll
  for (int i = 0; i < 8; ++i) {
    sqr[i]     = sqs[wave][hi * 8 + i];        // rows mi=0
    sqr[8 + i] = sqs[wave][16 + hi * 8 + i];   // rows mi=1
  }

  float rs[16];
#pragma unroll
  for (int i = 0; i < 8; ++i) {
    float r0 = sqr[i];
    float e0 = __expf(-0.5f * fmaxf(r0 + sqc0 - 2.0f * acc00[i], 0.0f));
    float e1 = __expf(-0.5f * fmaxf(r0 + sqc1 - 2.0f * acc01[i], 0.0f));
    rs[i] = e0 + e1;
    float r1 = sqr[8 + i];
    float f0 = __expf(-0.5f * fmaxf(r1 + sqc0 - 2.0f * acc10[i], 0.0f));
    float f1 = __expf(-0.5f * fmaxf(r1 + sqc1 - 2.0f * acc11[i], 0.0f));
    rs[8 + i] = f0 + f1;
  }
  // reduce over the 16 lanes of each half (columns within half share hi)
#pragma unroll
  for (int i = 0; i < 16; ++i) {
    float v = rs[i];
    v += __shfl_xor(v, 1, 32);
    v += __shfl_xor(v, 2, 32);
    v += __shfl_xor(v, 4, 32);
    v += __shfl_xor(v, 8, 32);
    rs[i] = v;
  }
  if (lm == 0) {                     // lanes 0 (rows hi=0) and 16 (rows hi=1)
    float* dst = density + (size_t)n * 32;
#pragma unroll
    for (int i = 0; i < 16; ++i) {
      int mi = i >> 3, r2 = i & 7;
      dst[mi * 16 + hi * 8 + r2] = rs[i] * (1.0f / 32.0f);
    }
  }
}

// ---------------------------------------------------------------------------
// Kernel 2: MLP + sigmoid + gamma -> attn[b][c][32][32].
// One wave per spatial position, 8 positions/wave; density gathered with
// async-to-LDS loads; weights staged (transposed) in LDS once per block.
// ---------------------------------------------------------------------------
__global__ __launch_bounds__(256) void kde_mlp_kernel(
    const float* __restrict__ density,
    const float* __restrict__ w1, const float* __restrict__ b1,
    const float* __restrict__ w2, const float* __restrict__ b2,
    float* __restrict__ attn) {
  __shared__ float w1t[128 * 32];   // [c][o]  (transposed: lane-major o)
  __shared__ float w2t[32 * 128];   // [o][o2] (transposed: lane-major o2)
  __shared__ float b1s[32], b2s[128];
  __shared__ float dbuf[8][128];
  __shared__ float ybuf[8][32];

  const int tid = threadIdx.x;
  for (int i = tid; i < 32 * 128; i += 256) {
    int o = i >> 7, c = i & 127;
    w1t[c * 32 + o] = w1[i];        // w1 is (32,128) row-major [o][c]
  }
  for (int i = tid; i < 128 * 32; i += 256) {
    int o2 = i >> 5, o = i & 31;
    w2t[o * 128 + o2] = w2[i];      // w2 is (128,32) row-major [o2][o]
  }
  if (tid < 32)  b1s[tid] = b1[tid];
  if (tid < 128) b2s[tid] = b2[tid];
  __syncthreads();

  const int wave = tid >> 5, lane = tid & 31;
  const int pbase = blockIdx.x * 64 + wave * 8;
  for (int it = 0; it < 8; ++it) {
    const int p  = pbase + it;
    const int b  = p >> 10;
    const int hy = (p >> 5) & 31;
    const int wx = p & 31;
#pragma unroll
    for (int g = 0; g < 4; ++g) {
      size_t nidx = ((size_t)((b * 4 + g) * 32 + hy) * 32 + wx);
      async_b32((unsigned int)(uintptr_t)&dbuf[wave][g * 32 + lane],
                density + nidx * 32 + lane);
    }
    asm volatile("s_wait_asynccnt 0x0" ::: "memory");
    __syncthreads();
    float y = b1s[lane];
#pragma unroll 8
    for (int c = 0; c < 128; ++c) y = fmaf(w1t[c * 32 + lane], dbuf[wave][c], y);
    ybuf[wave][lane] = fmaxf(y, 0.0f);
    __syncthreads();
#pragma unroll
    for (int g = 0; g < 4; ++g) {
      int o2 = g * 32 + lane;
      float z = b2s[o2];
#pragma unroll
      for (int o = 0; o < 32; ++o) z = fmaf(w2t[o * 128 + o2], ybuf[wave][o], z);
      float a = 1.0f / (1.0f + __expf(-z)) + 0.5f;   // sigmoid + GAMMA
      attn[(((size_t)b * 128 + o2) * 32 + hy) * 32 + wx] = a;
    }
    __syncthreads();
  }
}

// ---------------------------------------------------------------------------
// Kernel 3: out = x * bilinear_upsample(attn, 8x, align-corners).
// float4 per thread; attn (4MB) is L2-resident on MI455X (192MB L2).
// ---------------------------------------------------------------------------
__global__ __launch_bounds__(256) void kde_apply_kernel(
    const float* __restrict__ x, const float* __restrict__ attn,
    float* __restrict__ out) {
  const size_t idx  = (size_t)blockIdx.x * 256 + threadIdx.x;
  const size_t base = idx * 4;
  const int xcol = (int)(base & 255);
  const int y    = (int)((base >> 8) & 255);
  const size_t bc = base >> 16;              // b*128 + ch
  const float SCL = 31.0f / 255.0f;

  float yi = (float)y * SCL;
  int y0 = (int)yi; if (y0 > 30) y0 = 30;
  float wy = yi - (float)y0;

  const float* a0 = attn + (bc * 32 + (size_t)y0) * 32;
  const float* a1 = a0 + 32;
  float4 xin = *(const float4*)(x + base);
  float xv[4] = {xin.x, xin.y, xin.z, xin.w};
  float r[4];
#pragma unroll
  for (int k = 0; k < 4; ++k) {
    float xi2 = (float)(xcol + k) * SCL;
    int x0 = (int)xi2; if (x0 > 30) x0 = 30;
    float wxv = xi2 - (float)x0;
    float c0 = a0[x0]     * (1.0f - wy) + a1[x0]     * wy;
    float c1 = a0[x0 + 1] * (1.0f - wy) + a1[x0 + 1] * wy;
    r[k] = xv[k] * (c0 * (1.0f - wxv) + c1 * wxv);
  }
  float4 o = {r[0], r[1], r[2], r[3]};
  *(float4*)(out + base) = o;
}

// ---------------------------------------------------------------------------
extern "C" void kernel_launch(void* const* d_in, const int* in_sizes, int n_in,
                              void* d_out, int out_size, void* d_ws, size_t ws_size,
                              hipStream_t stream) {
  const float* x  = (const float*)d_in[0];   // (8,128,256,256)
  const float* w1 = (const float*)d_in[1];   // (32,128)
  const float* b1 = (const float*)d_in[2];   // (32,)
  const float* w2 = (const float*)d_in[3];   // (128,32)
  const float* b2 = (const float*)d_in[4];   // (128,)
  float* out = (float*)d_out;

  float* density = (float*)d_ws;                         // 32768*32 f32 = 4MB
  float* attn    = density + (size_t)32768 * 32;         // 8*128*32*32 = 4MB

  kde_density_kernel<<<8192, 128, 0, stream>>>(x, density);
  kde_mlp_kernel<<<128, 256, 0, stream>>>(density, w1, b1, w2, b2, attn);
  kde_apply_kernel<<<65536, 256, 0, stream>>>(x, attn, out);
}
